// ValueNetwork_68453188764142
// MI455X (gfx1250) — matrix operand order repaired and checked by reading, not compiled, not run on hardware
//
#include <hip/hip_runtime.h>
#include <hip/hip_bf16.h>

// ---------------------------------------------------------------------------
// CDNA5 (gfx1250) fused value-network, wave32 + v_wmma_f32_16x16x32_f16.
// ---------------------------------------------------------------------------

typedef __attribute__((ext_vector_type(16))) _Float16 v16h;
typedef __attribute__((ext_vector_type(8)))  float    v8f;

#define DEV static __device__ __forceinline__

// ---- workspace layout (units: _Float16 halves; all offsets 32B aligned) ----
// Fragment-packed B matrices: per 32(K)x16(N) tile, 32 lanes x 16 halves = 512
// halves (1 KB). Tile order: [ntile][ktile][lane][e].
static constexpr size_t OFF_W0H  = 0;        // 32x256  (K pad from 7)  : 8192
static constexpr size_t OFF_W0O  = 8192;     // 32x256  (K pad from 7)
static constexpr size_t OFF_W0R  = 16384;    // 32x256  (K pad from 6)
static constexpr size_t OFF_W1H  = 24576;    // 256x128 : 32768
static constexpr size_t OFF_W1O  = 57344;
static constexpr size_t OFF_W1R  = 90112;
static constexpr size_t OFF_WC1H = 122880;   // 128x32 combined : 4096
static constexpr size_t OFF_WC1O = 126976;
static constexpr size_t OFF_WC1R = 131072;
static constexpr size_t OFF_R1P  = 135168;   // 8 x (128x32) c1_relW : 32768
static constexpr size_t OFF_C2A  = 167936;   // 32x128 c2_relW[1] : 4096
static constexpr size_t OFF_C2B  = 172032;   // 32x128 c2_relW[2]
static constexpr size_t OFF_C2R  = 176128;   // 32x128 c2_rootW[1]+[2]
static constexpr size_t OFF_VW0  = 180224;   // 128x256 : 32768
static constexpr size_t OFF_VW1  = 212992;   // 256x128 : 32768
static constexpr size_t OFF_ACT  = 245760;   // activations start here

// ---- WMMA helpers ---------------------------------------------------------

DEV v8f wmma16(v16h a, v16h b, v8f c) {
  // (neg_a, A, neg_b, B, c_mod, C, reuse_a, reuse_b)
  return __builtin_amdgcn_wmma_f32_16x16x32_f16(false, a, false, b,
                                                (short)0, c, false, false);
}

// A fragment (16xK row-major f16 in LDS) -> v16h in contiguous k order (same as pack_b)
// lane L: m = L&15, half = L>>4; element e: K = half*16 + e.
// Pairs are K-consecutive -> ds_load_b32 pairs.
DEV v16h load_a_frag(const _Float16* A, int lda, int k0, int lane) {
  const int m = lane & 15, half = lane >> 4;
  const _Float16* row = A + m * lda + k0;
  v16h a;
#pragma unroll
  for (int j = 0; j < 8; ++j) {
    const int k = (half << 4) + (j << 1);
    a[2 * j]     = row[k];
    a[2 * j + 1] = row[k + 1];
  }
  return a;
}

// B fragment from fragment-packed weights: contiguous 32 bytes per lane.
DEV v16h load_b_frag(const _Float16* pack, int frag, int lane) {
  const v16h* p = reinterpret_cast<const v16h*>(pack + ((size_t)frag << 9));
  return p[lane];
}

// ---- k_prep: pack weights into fragment layout ----------------------------

DEV void pack_b(_Float16* dst, int Kreal, int Kpad, int N, const float* W,
                int tid, int stride) {
  const int kt = Kpad >> 5;
  const int total = (N >> 4) * kt * 512;
  for (int i = tid; i < total; i += stride) {
    const int frag = i >> 9, idx = i & 511;
    const int lane = idx >> 4, e = idx & 15;
    const int ntile = frag / kt, ktile = frag - ntile * kt;
    const int n = (ntile << 4) + (lane & 15);
    const int k = (ktile << 5) + ((lane >> 4) << 4) + e;
    const float v = (k < Kreal) ? W[(size_t)k * N + n] : 0.0f;
    dst[i] = (_Float16)v;
  }
}

DEV void pack_b_sum(_Float16* dst, int K, int N, const float* Wa,
                    const float* Wb, const float* Wc, const float* Wd,
                    float sd, int tid, int stride) {
  const int kt = K >> 5;
  const int total = (N >> 4) * kt * 512;
  for (int i = tid; i < total; i += stride) {
    const int frag = i >> 9, idx = i & 511;
    const int lane = idx >> 4, e = idx & 15;
    const int ntile = frag / kt, ktile = frag - ntile * kt;
    const int n = (ntile << 4) + (lane & 15);
    const size_t kn =
        (size_t)((ktile << 5) + ((lane >> 4) << 4) + e) * N + n;
    float v = Wa[kn] + Wb[kn];
    if (Wc) v += Wc[kn];
    if (Wd) v += sd * Wd[kn];
    dst[i] = (_Float16)v;
  }
}

__global__ void k_prep(const float* wr_W0, const float* wr_W1,
                       const float* wh_W0, const float* wh_W1,
                       const float* wo_W0, const float* wo_W1,
                       const float* c1_relW, const float* c1_rootW,
                       const float* c2_relW, const float* c2_rootW,
                       const float* v_W0, const float* v_W1, _Float16* ws) {
  const int tid = blockIdx.x * blockDim.x + threadIdx.x;
  const int stride = gridDim.x * blockDim.x;
  pack_b(ws + OFF_W0H, 7, 32, 256, wh_W0, tid, stride);
  pack_b(ws + OFF_W0O, 7, 32, 256, wo_W0, tid, stride);
  pack_b(ws + OFF_W0R, 6, 32, 256, wr_W0, tid, stride);
  pack_b(ws + OFF_W1H, 256, 256, 128, wh_W1, tid, stride);
  pack_b(ws + OFF_W1O, 256, 256, 128, wo_W1, tid, stride);
  pack_b(ws + OFF_W1R, 256, 256, 128, wr_W1, tid, stride);
  // combined conv1 per-agent weights (each source matrix is 128*32 = 4096)
  pack_b_sum(ws + OFF_WC1H, 128, 32, c1_rootW + 0 * 4096, c1_rootW + 4 * 4096,
             c1_rootW + 6 * 4096, c1_relW + 6 * 4096, -1.f, tid, stride);
  pack_b_sum(ws + OFF_WC1O, 128, 32, c1_rootW + 3 * 4096, c1_rootW + 5 * 4096,
             c1_rootW + 7 * 4096, c1_relW + 7 * 4096, -1.f, tid, stride);
  pack_b_sum(ws + OFF_WC1R, 128, 32, c1_rootW + 1 * 4096, c1_rootW + 2 * 4096,
             nullptr, nullptr, 0.f, tid, stride);
  for (int i = 0; i < 8; ++i)
    pack_b(ws + OFF_R1P + (size_t)i * 4096, 128, 128, 32,
           c1_relW + (size_t)i * 4096, tid, stride);
  pack_b(ws + OFF_C2A, 32, 32, 128, c2_relW + 1 * 4096, tid, stride);
  pack_b(ws + OFF_C2B, 32, 32, 128, c2_relW + 2 * 4096, tid, stride);
  pack_b_sum(ws + OFF_C2R, 32, 128, c2_rootW + 1 * 4096, c2_rootW + 2 * 4096,
             nullptr, nullptr, 0.f, tid, stride);
  pack_b(ws + OFF_VW0, 128, 128, 256, v_W0, tid, stride);
  pack_b(ws + OFF_VW1, 256, 256, 128, v_W1, tid, stride);
}

// ---- k_embed: per-16-batch tile, 31 agent slots ---------------------------
// er/eh/eo 2-layer MLPs + combined conv1 per-agent term; accumulates sh1/so1.

__global__ __launch_bounds__(256) void k_embed(
    const float* __restrict__ state, int B, const _Float16* __restrict__ ws,
    const float* bh0, const float* bh1, const float* bo0, const float* bo1,
    const float* br0, const float* br1, _Float16* __restrict__ ER,
    _Float16* __restrict__ SH1, _Float16* __restrict__ SO1,
    _Float16* __restrict__ TH, _Float16* __restrict__ TO) {
  __shared__ _Float16 Xs[16][32];
  __shared__ _Float16 Hs[16][256];
  __shared__ _Float16 Es[16][128];
  __shared__ float SAs[2][16][128];  // [0]=sum_h, [1]=sum_o

  const int tid = threadIdx.x;
  const int wave = tid >> 5, lane = tid & 31;
  const int b0 = blockIdx.x * 16;

  for (int i = tid; i < 2 * 16 * 128; i += 256) ((float*)SAs)[i] = 0.f;
  __syncthreads();

  for (int slot = 0; slot < 31; ++slot) {
    const int type = (slot < 20) ? 0 : (slot < 30 ? 1 : 2);  // h / o / self
    // stage input rows (zero-padded to K=32)
    for (int i = tid; i < 512; i += 256) {
      const int r = i >> 5, c = i & 31;
      float v = 0.f;
      const int b = b0 + r;
      if (b < B) {
        if (type == 2) {
          if (c < 6) v = state[((size_t)b * 30 + 0) * 13 + c];
        } else {
          if (c < 7) v = state[((size_t)b * 30 + slot) * 13 + 6 + c];
        }
      }
      Xs[r][c] = (_Float16)v;
    }
    __syncthreads();

    const _Float16* W0 =
        ws + (type == 0 ? OFF_W0H : type == 1 ? OFF_W0O : OFF_W0R);
    const _Float16* W1 =
        ws + (type == 0 ? OFF_W1H : type == 1 ? OFF_W1O : OFF_W1R);
    const float* bias0 = (type == 0) ? bh0 : (type == 1 ? bo0 : br0);
    const float* bias1 = (type == 0) ? bh1 : (type == 1 ? bo1 : br1);

    // layer 1: 16x32 @ 32x256, 16 n-tiles over 8 waves
    {
      const v16h a = load_a_frag(&Xs[0][0], 32, 0, lane);
#pragma unroll
      for (int s = 0; s < 2; ++s) {
        const int nt = wave * 2 + s;
        const v16h b = load_b_frag(W0, nt, lane);
        v8f acc = {};
        acc = wmma16(a, b, acc);
        const int n = (nt << 4) + (lane & 15);
        const float bs = bias0[n];
#pragma unroll
        for (int i = 0; i < 8; ++i) {
          const int m = i + ((lane >> 4) << 3);
          Hs[m][n] = (_Float16)fmaxf(acc[i] + bs, 0.f);
        }
      }
    }
    __syncthreads();

    // layer 2: 16x256 @ 256x128, 8 n-tiles over 8 waves
    {
      const int nt = wave;
      v8f acc = {};
#pragma unroll
      for (int kt = 0; kt < 8; ++kt) {
        const v16h a = load_a_frag(&Hs[0][0], 256, kt << 5, lane);
        const v16h b = load_b_frag(W1, nt * 8 + kt, lane);
        acc = wmma16(a, b, acc);
      }
      const int n = (nt << 4) + (lane & 15);
      const float bs = bias1[n];
#pragma unroll
      for (int i = 0; i < 8; ++i) {
        const int m = i + ((lane >> 4) << 3);
        const float v = fmaxf(acc[i] + bs, 0.f);
        Es[m][n] = (_Float16)v;
        if (type < 2) {
          SAs[type][m][n] += v;  // columns are wave-private: no races
        } else if (b0 + m < B) {
          ER[(size_t)(b0 + m) * 128 + n] = (_Float16)v;
        }
      }
    }
    __syncthreads();

    // combined conv1 per-agent term: 16x128 @ 128x32 (2 n-tiles -> waves 0,1)
    if (type < 2 && wave < 2) {
      const _Float16* Wc = ws + (type == 0 ? OFF_WC1H : OFF_WC1O);
      const int nt = wave;
      v8f acc = {};
#pragma unroll
      for (int kt = 0; kt < 4; ++kt) {
        const v16h a = load_a_frag(&Es[0][0], 128, kt << 5, lane);
        const v16h b = load_b_frag(Wc, nt * 4 + kt, lane);
        acc = wmma16(a, b, acc);
      }
      const int n = (nt << 4) + (lane & 15);
#pragma unroll
      for (int i = 0; i < 8; ++i) {
        const int m = i + ((lane >> 4) << 3);
        if (b0 + m < B) {
          if (type == 0)
            TH[((size_t)(b0 + m) * 20 + slot) * 32 + n] = (_Float16)acc[i];
          else
            TO[((size_t)(b0 + m) * 10 + (slot - 20)) * 32 + n] =
                (_Float16)acc[i];
        }
      }
    }
    __syncthreads();
  }

  for (int i = tid; i < 16 * 128; i += 256) {
    const int r = i >> 7, c = i & 127;
    if (b0 + r < B) {
      SH1[(size_t)(b0 + r) * 128 + c] = (_Float16)SAs[0][r][c];
      SO1[(size_t)(b0 + r) * 128 + c] = (_Float16)SAs[1][r][c];
    }
  }
}

// ---- k_final: conv consts + agent sums + conv2 root + value head ----------

__global__ __launch_bounds__(256) void k_final(
    int B, const _Float16* __restrict__ ws, const _Float16* __restrict__ ER,
    const _Float16* __restrict__ SH1, const _Float16* __restrict__ SO1,
    const _Float16* __restrict__ TH, const _Float16* __restrict__ TO,
    const float* __restrict__ c1_relb, const float* __restrict__ c2_relb,
    const float* __restrict__ vb0, const float* __restrict__ vb1,
    const float* __restrict__ vW2, const float* __restrict__ vb2,
    float* __restrict__ out) {
  __shared__ _Float16 ERs[16][128], SHs[16][128], SOs[16][128];
  __shared__ float CH[16][32], CO[16][32];
  __shared__ _Float16 R1s[16][32], SH2s[16][32], SO2s[16][32];
  __shared__ _Float16 HRs[16][128];
  __shared__ _Float16 V1s[16][256];
  __shared__ _Float16 V2s[16][128];

  const int tid = threadIdx.x, wave = tid >> 5, lane = tid & 31;
  const int b0 = blockIdx.x * 16;

  for (int i = tid; i < 16 * 128; i += 256) {
    const int r = i >> 7, c = i & 127;
    const size_t g = (size_t)(b0 + r) * 128 + c;
    const bool ok = (b0 + r) < B;
    ERs[r][c] = ok ? ER[g] : (_Float16)0.f;
    SHs[r][c] = ok ? SH1[g] : (_Float16)0.f;
    SOs[r][c] = ok ? SO1[g] : (_Float16)0.f;
  }
  __syncthreads();

  // conv1 broadcast terms: groups {const_h, const_o, out_r1}, each 3 GEMMs
  // of 16x128 @ 128x32, 2 n-tiles -> waves 0..5.
  if (wave < 6) {
    const int group = wave >> 1;
    const int nt = wave & 1;
    const _Float16* srcs[3];
    const _Float16* packs[3];
    int bi0, bi1, bi2;
    if (group == 0) {  // const_h = er@relW0 + so@relW4 + sh@relW6 (+biases)
      srcs[0] = &ERs[0][0]; packs[0] = ws + OFF_R1P + 0 * 4096;
      srcs[1] = &SOs[0][0]; packs[1] = ws + OFF_R1P + 4 * 4096;
      srcs[2] = &SHs[0][0]; packs[2] = ws + OFF_R1P + 6 * 4096;
      bi0 = 0; bi1 = 4; bi2 = 6;
    } else if (group == 1) {  // const_o = er@relW3 + sh@relW5 + so@relW7
      srcs[0] = &ERs[0][0]; packs[0] = ws + OFF_R1P + 3 * 4096;
      srcs[1] = &SHs[0][0]; packs[1] = ws + OFF_R1P + 5 * 4096;
      srcs[2] = &SOs[0][0]; packs[2] = ws + OFF_R1P + 7 * 4096;
      bi0 = 3; bi1 = 5; bi2 = 7;
    } else {  // out_r1 = sh@relW1 + so@relW2 + er@(rootW1+rootW2)
      srcs[0] = &SHs[0][0]; packs[0] = ws + OFF_R1P + 1 * 4096;
      srcs[1] = &SOs[0][0]; packs[1] = ws + OFF_R1P + 2 * 4096;
      srcs[2] = &ERs[0][0]; packs[2] = ws + OFF_WC1R;
      bi0 = 1; bi1 = 2; bi2 = -1;
    }
    v8f acc = {};
    for (int t = 0; t < 3; ++t) {
#pragma unroll
      for (int kt = 0; kt < 4; ++kt) {
        const v16h a = load_a_frag(srcs[t], 128, kt << 5, lane);
        const v16h b = load_b_frag(packs[t], nt * 4 + kt, lane);
        acc = wmma16(a, b, acc);
      }
    }
    const int n = (nt << 4) + (lane & 15);
    const float bias = c1_relb[bi0 * 32 + n] + c1_relb[bi1 * 32 + n] +
                       (bi2 >= 0 ? c1_relb[bi2 * 32 + n] : 0.f);
#pragma unroll
    for (int i = 0; i < 8; ++i) {
      const int m = i + ((lane >> 4) << 3);
      const float v = acc[i] + bias;
      if (group == 0)
        CH[m][n] = v;
      else if (group == 1)
        CO[m][n] = v;
      else
        R1s[m][n] = (_Float16)fmaxf(v, 0.f);
    }
  }
  __syncthreads();

  // sh2 / so2: elementwise relu-sum over agents
  for (int i = tid; i < 512; i += 256) {
    const int m = i >> 5, c = i & 31;
    const bool ok = (b0 + m) < B;
    const float ch = CH[m][c];
    float s = 0.f;
    if (ok)
      for (int n = 0; n < 20; ++n)
        s += fmaxf(ch + (float)TH[((size_t)(b0 + m) * 20 + n) * 32 + c], 0.f);
    SH2s[m][c] = (_Float16)s;
    const float co = CO[m][c];
    float s2 = 0.f;
    if (ok)
      for (int n = 0; n < 10; ++n)
        s2 += fmaxf(co + (float)TO[((size_t)(b0 + m) * 10 + n) * 32 + c], 0.f);
    SO2s[m][c] = (_Float16)s2;
  }
  __syncthreads();

  // conv2 root row: out_r2 = sh2@c2relW1 + so2@c2relW2 + r1@(rootW1+rootW2)
  {
    const int nt = wave;  // N=128 -> 8 n-tiles
    v8f acc = {};
    acc = wmma16(load_a_frag(&SH2s[0][0], 32, 0, lane),
                 load_b_frag(ws + OFF_C2A, nt, lane), acc);
    acc = wmma16(load_a_frag(&SO2s[0][0], 32, 0, lane),
                 load_b_frag(ws + OFF_C2B, nt, lane), acc);
    acc = wmma16(load_a_frag(&R1s[0][0], 32, 0, lane),
                 load_b_frag(ws + OFF_C2R, nt, lane), acc);
    const int n = (nt << 4) + (lane & 15);
    const float bias = c2_relb[128 + n] + c2_relb[256 + n];
#pragma unroll
    for (int i = 0; i < 8; ++i) {
      const int m = i + ((lane >> 4) << 3);
      HRs[m][n] = (_Float16)fmaxf(acc[i] + bias, 0.f);
    }
  }
  __syncthreads();

  // value layer 0: 16x128 @ 128x256
  for (int s = 0; s < 2; ++s) {
    const int nt = wave + s * 8;
    v8f acc = {};
#pragma unroll
    for (int kt = 0; kt < 4; ++kt) {
      const v16h a = load_a_frag(&HRs[0][0], 128, kt << 5, lane);
      const v16h b = load_b_frag(ws + OFF_VW0, nt * 4 + kt, lane);
      acc = wmma16(a, b, acc);
    }
    const int n = (nt << 4) + (lane & 15);
    const float bs = vb0[n];
#pragma unroll
    for (int i = 0; i < 8; ++i) {
      const int m = i + ((lane >> 4) << 3);
      V1s[m][n] = (_Float16)fmaxf(acc[i] + bs, 0.f);
    }
  }
  __syncthreads();

  // value layer 1: 16x256 @ 256x128
  {
    const int nt = wave;
    v8f acc = {};
#pragma unroll
    for (int kt = 0; kt < 8; ++kt) {
      const v16h a = load_a_frag(&V1s[0][0], 256, kt << 5, lane);
      const v16h b = load_b_frag(ws + OFF_VW1, nt * 8 + kt, lane);
      acc = wmma16(a, b, acc);
    }
    const int n = (nt << 4) + (lane & 15);
    const float bs = vb1[n];
#pragma unroll
    for (int i = 0; i < 8; ++i) {
      const int m = i + ((lane >> 4) << 3);
      V2s[m][n] = (_Float16)fmaxf(acc[i] + bs, 0.f);
    }
  }
  __syncthreads();

  // value layer 2: 16x128 @ 128x1 (tiny dot per row)
  if (tid < 16) {
    const int m = tid;
    float s = 0.f;
    for (int k = 0; k < 128; ++k) s += (float)V2s[m][k] * vW2[k];
    if (b0 + m < B) out[b0 + m] = s + vb2[0];
  }
}

// ---- host-side launch -----------------------------------------------------

extern "C" void kernel_launch(void* const* d_in, const int* in_sizes, int n_in,
                              void* d_out, int out_size, void* d_ws,
                              size_t ws_size, hipStream_t stream) {
  (void)n_in; (void)out_size; (void)ws_size;
  const float* state    = (const float*)d_in[0];
  const float* wr_W0    = (const float*)d_in[2];
  const float* wr_b0    = (const float*)d_in[3];
  const float* wr_W1    = (const float*)d_in[4];
  const float* wr_b1    = (const float*)d_in[5];
  const float* wh_W0    = (const float*)d_in[6];
  const float* wh_b0    = (const float*)d_in[7];
  const float* wh_W1    = (const float*)d_in[8];
  const float* wh_b1    = (const float*)d_in[9];
  const float* wo_W0    = (const float*)d_in[10];
  const float* wo_b0    = (const float*)d_in[11];
  const float* wo_W1    = (const float*)d_in[12];
  const float* wo_b1    = (const float*)d_in[13];
  const float* c1_relW  = (const float*)d_in[14];
  const float* c1_relb  = (const float*)d_in[15];
  const float* c1_rootW = (const float*)d_in[16];
  const float* c2_relW  = (const float*)d_in[17];
  const float* c2_relb  = (const float*)d_in[18];
  const float* c2_rootW = (const float*)d_in[19];
  const float* v_W0     = (const float*)d_in[20];
  const float* v_b0     = (const float*)d_in[21];
  const float* v_W1     = (const float*)d_in[22];
  const float* v_b1     = (const float*)d_in[23];
  const float* v_W2     = (const float*)d_in[24];
  const float* v_b2     = (const float*)d_in[25];

  const int B = in_sizes[0] / (30 * 13);
  _Float16* ws = (_Float16*)d_ws;
  _Float16* ER  = ws + OFF_ACT;
  _Float16* SH1 = ER + (size_t)B * 128;
  _Float16* SO1 = SH1 + (size_t)B * 128;
  _Float16* TH  = SO1 + (size_t)B * 128;
  _Float16* TO  = TH + (size_t)B * 20 * 32;

  k_prep<<<dim3(120), dim3(256), 0, stream>>>(
      wr_W0, wr_W1, wh_W0, wh_W1, wo_W0, wo_W1, c1_relW, c1_rootW, c2_relW,
      c2_rootW, v_W0, v_W1, ws);

  const int nb = (B + 15) / 16;
  k_embed<<<dim3(nb), dim3(256), 0, stream>>>(
      state, B, ws, wh_b0, wh_b1, wo_b0, wo_b1, wr_b0, wr_b1, ER, SH1, SO1,
      TH, TO);
  k_final<<<dim3(nb), dim3(256), 0, stream>>>(
      B, ws, ER, SH1, SO1, TH, TO, c1_relb, c2_relb, v_b0, v_b1, v_W2, v_b2,
      (float*)d_out);
}
